// EmbNet_30949534335493
// MI455X (gfx1250) — matrix operand order repaired and checked by reading, not compiled, not run on hardware
//
#include <hip/hip_runtime.h>
#include <hip/hip_bf16.h>

// ---------------------------------------------------------------------------
// GATv2 GNN for MI455X (gfx1250, wave32).  All dense GEMMs run on
// v_wmma_f32_16x16x32_f16 (f16 in, f32 accumulate).  Tiles are fully staged
// in LDS (320 KB/WGP) so the WMMA chain runs barrier-free; sched_group_barrier
// pins the schedule to [18 ds_load] -> [8 v_wmma] per k-step so the waits are
// partial (dscnt<=14,12,...) instead of a full dscnt==0 stall per WMMA.
// Scatter/softmax parts are f32 VALU + global atomics (L2-resident gathers).
// ---------------------------------------------------------------------------

#define NN   20000          // nodes
#define NE   320000         // edges
#define NR   (NN + NE)      // attention rows (edges + self loops)
#define UDIM 128            // feature width
#define NDEPTH 4
#define EPSV 1e-5f

typedef __attribute__((ext_vector_type(16))) _Float16 v16h;
typedef __attribute__((ext_vector_type(8)))  _Float16 v8h;
typedef __attribute__((ext_vector_type(8)))  float    v8f;

__device__ __forceinline__ float silu_f(float v) { return v / (1.0f + __expf(-v)); }

// monotonic float <-> uint key for atomicMax-based segment max
__device__ __forceinline__ unsigned fkey(float f) {
    unsigned u = __float_as_uint(f);
    return (u & 0x80000000u) ? ~u : (u | 0x80000000u);
}
__device__ __forceinline__ float funkey(unsigned k) {
    return (k & 0x80000000u) ? __uint_as_float(k & 0x7FFFFFFFu)
                             : __uint_as_float(~k);
}

// ---------------------------------------------------------------------------
// One K-step (K=32) of a 16x128 output strip for one wave.
// sA row-major [row][k] (stride sa), sB transposed [col][k] (stride sb).
// A fragment per ISA 7.12.2 (interleaved 8-blocks); B fragment same interleaved
// 8-blocks.  sched_group_barrier enforces: all 18 DS reads first, then the
// 8-WMMA chain (incremental dscnt waits -> pipelined LDS latency).
// ---------------------------------------------------------------------------
__device__ __forceinline__ void wmma_ktile(const _Float16* sA, int sa, int ka,
                                           const _Float16* sB, int sb, int kb,
                                           int lane, int wave, v8f acc[8]) {
    const int arow  = wave * 16 + (lane & 15);
    const int akoff = ka + ((lane >> 4) << 3);        // +0 or +8
    v8h alo = *(const v8h*)(sA + arow * sa + akoff);
    v8h ahi = *(const v8h*)(sA + arow * sa + akoff + 16);
    v16h afr;
#pragma unroll
    for (int i = 0; i < 8; ++i) { afr[i] = alo[i]; afr[i + 8] = ahi[i]; }

    const int bkoff = kb + ((lane >> 4) << 3);        // +0 or +8
    v16h bfr[8];
#pragma unroll
    for (int nt = 0; nt < 8; ++nt) {
        const int bcol = nt * 16 + (lane & 15);
        v8h blo = *(const v8h*)(sB + bcol * sb + bkoff);
        v8h bhi = *(const v8h*)(sB + bcol * sb + bkoff + 16);
#pragma unroll
        for (int i = 0; i < 8; ++i) { bfr[nt][i] = blo[i]; bfr[nt][i + 8] = bhi[i]; }
    }
#pragma unroll
    for (int nt = 0; nt < 8; ++nt)
        acc[nt] = __builtin_amdgcn_wmma_f32_16x16x32_f16(
            false, afr, false, bfr[nt], (short)0, acc[nt], false, false);

#if __has_builtin(__builtin_amdgcn_sched_group_barrier)
    // desired schedule for this k-step: 18 DS reads (2 A + 16 B), then 8 WMMA
    __builtin_amdgcn_sched_group_barrier(0x100, 18, 0);  // DS read group
    __builtin_amdgcn_sched_group_barrier(0x008, 8, 0);   // MFMA/WMMA group
#endif
}

template <int ACT, bool HB>
__device__ __forceinline__ void gat_epilogue(float* __restrict__ C, const v8f acc[8],
                                             const float* __restrict__ bias,
                                             int rowBase, int lane, int wave, int M) {
    const int colLane = lane & 15;
    const int rowHalf = (lane >> 4) * 8;
#pragma unroll
    for (int nt = 0; nt < 8; ++nt) {
        const int col = nt * 16 + colLane;
        const float bv = HB ? bias[col] : 0.0f;
#pragma unroll
        for (int r = 0; r < 8; ++r) {
            const int grow = rowBase + wave * 16 + rowHalf + r;
            if (grow < M) {
                float v = acc[nt][r] + bv;
                if (ACT == 1) v = silu_f(v);
                C[(size_t)grow * 128 + col] = v;
            }
        }
    }
}

// ---------------------------------------------------------------------------
// GEMM (K = 128): C[M x 128] = act(A[M x 128] @ B[128 x 128] + bias)
// Full A tile (32 KB) + full B (32 KB) staged once; ONE barrier per block,
// then 32 back-to-back WMMAs per wave.
// ---------------------------------------------------------------------------
template <int ACT, bool HB>
__global__ __launch_bounds__(256) void gat_gemm128_kernel(
    const float* __restrict__ A, const _Float16* __restrict__ B,
    const float* __restrict__ bias, float* __restrict__ C, int M) {
    __shared__ __align__(16) _Float16 sA[128 * 128];
    __shared__ __align__(16) _Float16 sB[128 * 128];   // [col][k]
    const int tid  = threadIdx.x;
    const int lane = tid & 31;
    const int wave = tid >> 5;
    const int rowBlock = blockIdx.x * 128;

    for (int i = tid; i < 128 * 128; i += 256) {       // B transposed
        const int c = i >> 7, k = i & 127;
        sB[c * 128 + k] = B[(size_t)k * 128 + c];
    }
    for (int i = tid; i < 128 * 128; i += 256) {       // A f32 -> f16
        const int r = i >> 7, k = i & 127;
        const int gr = rowBlock + r;
        sA[i] = (_Float16)((gr < M) ? A[(size_t)gr * 128 + k] : 0.0f);
    }
    __syncthreads();

    v8f acc[8] = {};
#pragma unroll
    for (int k0 = 0; k0 < 128; k0 += 32)
        wmma_ktile(sA, 128, k0, sB, 128, k0, lane, wave, acc);

    gat_epilogue<ACT, HB>(C, acc, bias, rowBlock, lane, wave, M);
}

// ---------------------------------------------------------------------------
// Edge MLP layer-1 GEMM, gathered A = [h[row] | h[col] | w0], K = 384.
// Full B (96 KB) resident in LDS for the whole block; A staged per k-step.
// C[E x 128] = silu(A @ W1 + b1)
// ---------------------------------------------------------------------------
__global__ __launch_bounds__(256) void gat_edge_mlp1_kernel(
    const float* __restrict__ h, const float* __restrict__ w0,
    const int* __restrict__ rowi, const int* __restrict__ coli,
    const _Float16* __restrict__ B, const float* __restrict__ bias,
    float* __restrict__ C) {
    __shared__ __align__(16) _Float16 sB[128 * 384];   // [col][k], 96 KB
    __shared__ __align__(16) _Float16 sA[128 * 32];    // per-kstep slice, 8 KB
    __shared__ int sSrc[128];
    __shared__ int sDst[128];
    const int tid  = threadIdx.x;
    const int lane = tid & 31;
    const int wave = tid >> 5;
    const int eBase = blockIdx.x * 128;

    if (tid < 128) {
        const int ge = eBase + tid;
        sSrc[tid] = (ge < NE) ? rowi[ge] : 0;
        sDst[tid] = (ge < NE) ? coli[ge] : 0;
    }
    for (int i = tid; i < 128 * 384; i += 256) {       // B transposed
        const int c = i / 384, k = i - c * 384;
        sB[c * 384 + k] = B[(size_t)k * 128 + c];
    }

    v8f acc[8] = {};
    for (int k0 = 0; k0 < 384; k0 += 32) {
        __syncthreads();                                // sA reuse / first-iter sB,sSrc
        const int region = k0 >> 7;                     // 0: h[row], 1: h[col], 2: w0
        const int lk     = k0 & 127;
        for (int i = tid; i < 128 * 32; i += 256) {
            const int r = i >> 5, kk = i & 31;
            const int ge = eBase + r;
            float v = 0.0f;
            if (ge < NE) {
                if (region == 0)      v = h[(size_t)sSrc[r] * UDIM + lk + kk];
                else if (region == 1) v = h[(size_t)sDst[r] * UDIM + lk + kk];
                else                  v = w0[(size_t)ge * UDIM + lk + kk];
            }
            sA[i] = (_Float16)v;
        }
        __syncthreads();
        wmma_ktile(sA, 32, 0, sB, 384, k0, lane, wave, acc);
    }

    gat_epilogue<1, true>(C, acc, bias, eBase, lane, wave, NE);
}

// ---------------------------------------------------------------------------
// Elementwise / scatter helpers
// ---------------------------------------------------------------------------
__global__ void gat_zero_kernel(float* __restrict__ p, long n) {
    long i = (long)blockIdx.x * blockDim.x + threadIdx.x;
    if (i < n) p[i] = 0.0f;
}

__global__ void gat_copy_kernel(const float* __restrict__ s, float* __restrict__ d, long n) {
    long i = (long)blockIdx.x * blockDim.x + threadIdx.x;
    if (i < n) d[i] = s[i];
}

__global__ void gat_cvt_f16_kernel(const float* __restrict__ s, _Float16* __restrict__ d, long n) {
    long i = (long)blockIdx.x * blockDim.x + threadIdx.x;
    if (i < n) d[i] = (_Float16)s[i];
}

__global__ void gat_init_bias_kernel(float* __restrict__ x, const float* __restrict__ b, long rows) {
    long i = (long)blockIdx.x * blockDim.x + threadIdx.x;
    if (i < rows * UDIM) x[i] = b[i & 127];
}

// h = silu(x @ W[2x128] + b)
__global__ void gat_embed_node_kernel(const float* __restrict__ x, const float* __restrict__ W,
                                      const float* __restrict__ b, float* __restrict__ h) {
    long i = (long)blockIdx.x * blockDim.x + threadIdx.x;
    if (i < (long)NN * UDIM) {
        const int n = (int)(i >> 7), c = (int)(i & 127);
        float v = x[n * 2] * W[c] + x[n * 2 + 1] * W[128 + c] + b[c];
        h[i] = silu_f(v);
    }
}

// w = silu(edge_attr @ W[1x128] + b)
__global__ void gat_embed_edge_kernel(const float* __restrict__ ea, const float* __restrict__ W,
                                      const float* __restrict__ b, float* __restrict__ w) {
    long i = (long)blockIdx.x * blockDim.x + threadIdx.x;
    if (i < (long)NE * UDIM) {
        const int e = (int)(i >> 7), c = (int)(i & 127);
        w[i] = silu_f(ea[e] * W[c] + b[c]);
    }
}

__global__ void gat_deg_kernel(const int* __restrict__ coli, float* __restrict__ deg) {
    long i = (long)blockIdx.x * blockDim.x + threadIdx.x;
    if (i < NE) atomicAdd(&deg[coli[i]], 1.0f);
}

__global__ void gat_loop_acc_kernel(const float* __restrict__ w, const int* __restrict__ coli,
                                    float* __restrict__ lpat) {
    long i = (long)blockIdx.x * blockDim.x + threadIdx.x;
    if (i < (long)NE * UDIM) {
        const int e = (int)(i >> 7), c = (int)(i & 127);
        atomicAdd(&lpat[(size_t)coli[e] * UDIM + c], w[i]);
    }
}

__global__ void gat_loop_div_kernel(float* __restrict__ lpat, const float* __restrict__ deg) {
    long i = (long)blockIdx.x * blockDim.x + threadIdx.x;
    if (i < (long)NN * UDIM) {
        float d = deg[i >> 7];
        lpat[i] /= (d < 1.0f ? 1.0f : d);
    }
}

// per (row, head) score: e = sum_c leaky(xl[src]+xr[dst]+ef) * att
__global__ void gat_attn_score_kernel(const float* __restrict__ xl, const float* __restrict__ xr,
                                      const float* __restrict__ ef, const int* __restrict__ rowi,
                                      const int* __restrict__ coli, const float* __restrict__ att,
                                      float* __restrict__ esc, unsigned* __restrict__ mkey) {
    long idx = (long)blockIdx.x * blockDim.x + threadIdx.x;
    if (idx >= (long)NR * 4) return;
    const int r = (int)(idx >> 2), hh = (int)(idx & 3);
    const int src = (r < NE) ? rowi[r] : (r - NE);
    const int dst = (r < NE) ? coli[r] : (r - NE);
    const float* a0 = xl + (size_t)src * UDIM + hh * 32;
    const float* a1 = xr + (size_t)dst * UDIM + hh * 32;
    const float* a2 = ef + (size_t)r * UDIM + hh * 32;
    const float* at = att + hh * 32;
    float s = 0.0f;
#pragma unroll 8
    for (int c = 0; c < 32; ++c) {
        float z = a0[c] + a1[c] + a2[c];
        z = (z > 0.0f) ? z : 0.2f * z;
        s += z * at[c];
    }
    esc[idx] = s;
    atomicMax(&mkey[dst * 4 + hh], fkey(s));
}

__global__ void gat_attn_exp_kernel(float* __restrict__ esc, const unsigned* __restrict__ mkey,
                                    float* __restrict__ ssum, const int* __restrict__ rowi,
                                    const int* __restrict__ coli) {
    long idx = (long)blockIdx.x * blockDim.x + threadIdx.x;
    if (idx >= (long)NR * 4) return;
    const int r = (int)(idx >> 2), hh = (int)(idx & 3);
    const int dst = (r < NE) ? coli[r] : (r - NE);
    const float m = funkey(mkey[dst * 4 + hh]);
    const float a = __expf(esc[idx] - m);
    esc[idx] = a;
    atomicAdd(&ssum[dst * 4 + hh], a);
}

__global__ void gat_attn_agg_kernel(const float* __restrict__ esc, const float* __restrict__ ssum,
                                    const float* __restrict__ xl, const int* __restrict__ rowi,
                                    const int* __restrict__ coli, float* __restrict__ xnew) {
    long idx = (long)blockIdx.x * blockDim.x + threadIdx.x;
    if (idx >= (long)NR * UDIM) return;
    const int r = (int)(idx >> 7), c = (int)(idx & 127), hh = c >> 5;
    const int src = (r < NE) ? rowi[r] : (r - NE);
    const int dst = (r < NE) ? coli[r] : (r - NE);
    const float alpha = esc[(size_t)r * 4 + hh] / ssum[dst * 4 + hh];
    atomicAdd(&xnew[(size_t)dst * UDIM + c], alpha * xl[(size_t)src * UDIM + c]);
}

// column-wise sum / sumsq via LDS ds_add_f32 then one flush per block
__global__ __launch_bounds__(256) void gat_bn_stats_kernel(const float* __restrict__ X,
                                                           float* __restrict__ sums, int M) {
    __shared__ float ls[256];
    ls[threadIdx.x] = 0.0f;
    __syncthreads();
    const int c = threadIdx.x & 127;
    float s = 0.0f, s2 = 0.0f;
    for (int r = blockIdx.x * 2 + (threadIdx.x >> 7); r < M; r += gridDim.x * 2) {
        float v = X[(size_t)r * UDIM + c];
        s += v; s2 += v * v;
    }
    atomicAdd(&ls[c], s);
    atomicAdd(&ls[128 + c], s2);
    __syncthreads();
    atomicAdd(&sums[threadIdx.x], ls[threadIdx.x]);
}

// res += silu( (X - mu) * rsqrt(var+eps) * g + b )
__global__ void gat_bn_apply_kernel(const float* __restrict__ X, const float* __restrict__ sums,
                                    const float* __restrict__ g, const float* __restrict__ b,
                                    float* __restrict__ res, int M) {
    long idx = (long)blockIdx.x * blockDim.x + threadIdx.x;
    if (idx >= (long)M * UDIM) return;
    const int c = (int)(idx & 127);
    const float mu  = sums[c] / (float)M;
    const float var = sums[128 + c] / (float)M - mu * mu;
    const float v   = (X[idx] - mu) * rsqrtf(var + EPSV) * g[c] + b[c];
    res[idx] += silu_f(v);
}

// ---------------------------------------------------------------------------
extern "C" void kernel_launch(void* const* d_in, const int* in_sizes, int n_in,
                              void* d_out, int out_size, void* d_ws, size_t ws_size,
                              hipStream_t stream) {
    (void)in_sizes; (void)n_in; (void)out_size; (void)ws_size;

    const float* x     = (const float*)d_in[0];
    const int*   ei    = (const int*)d_in[1];
    const float* eattr = (const float*)d_in[2];
    const float* v0w   = (const float*)d_in[3];
    const float* v0b   = (const float*)d_in[4];
    const float* e0w   = (const float*)d_in[5];
    const float* e0b   = (const float*)d_in[6];
    const float* Wl    = (const float*)d_in[7];
    const float* bl    = (const float*)d_in[8];
    const float* Wr    = (const float*)d_in[9];
    const float* br    = (const float*)d_in[10];
    const float* We    = (const float*)d_in[11];
    const float* att   = (const float*)d_in[12];
    const float* gbias = (const float*)d_in[13];
    const float* W1    = (const float*)d_in[14];
    const float* b1    = (const float*)d_in[15];
    const float* W2    = (const float*)d_in[16];
    const float* b2    = (const float*)d_in[17];
    const float* vbng  = (const float*)d_in[18];
    const float* vbnb  = (const float*)d_in[19];
    const float* ebng  = (const float*)d_in[20];
    const float* ebnb  = (const float*)d_in[21];

    const int* rowi = ei;
    const int* coli = ei + NE;

    char* ws = (char*)d_ws;
    size_t off = 0;
    auto alloc = [&](size_t bytes) -> void* {
        void* p = ws + off;
        off += (bytes + 255) & ~(size_t)255;
        return p;
    };
    float*    h    = (float*)alloc((size_t)NN * UDIM * 4);
    float*    wbuf = (float*)alloc((size_t)NE * UDIM * 4);
    float*    bufA = (float*)alloc((size_t)NR * UDIM * 4);   // ef, then MLP hidden
    float*    bufB = (float*)alloc((size_t)NE * UDIM * 4);   // w_new
    float*    xl   = (float*)alloc((size_t)NN * UDIM * 4);
    float*    xr   = (float*)alloc((size_t)NN * UDIM * 4);
    float*    xnew = (float*)alloc((size_t)NN * UDIM * 4);
    float*    lpat = (float*)alloc((size_t)NN * UDIM * 4);
    float*    esc  = (float*)alloc((size_t)NR * 4 * 4);
    unsigned* mkey = (unsigned*)alloc((size_t)NN * 4 * 4);
    float*    ssum = (float*)alloc((size_t)NN * 4 * 4);
    float*    deg  = (float*)alloc((size_t)NN * 4);
    float*    bns  = (float*)alloc(256 * 4);
    _Float16* Wl_h = (_Float16*)alloc((size_t)UDIM * UDIM * 2);
    _Float16* Wr_h = (_Float16*)alloc((size_t)UDIM * UDIM * 2);
    _Float16* We_h = (_Float16*)alloc((size_t)UDIM * UDIM * 2);
    _Float16* W1_h = (_Float16*)alloc((size_t)3 * UDIM * UDIM * 2);
    _Float16* W2_h = (_Float16*)alloc((size_t)UDIM * UDIM * 2);

    const dim3 blk(256);
    auto g1 = [](long n) { return dim3((unsigned)((n + 255) / 256)); };
    const dim3 gNodeG((NN + 127) / 128);
    const dim3 gEdgeG((NE + 127) / 128);

    gat_embed_node_kernel<<<g1((long)NN * UDIM), blk, 0, stream>>>(x, v0w, v0b, h);
    gat_embed_edge_kernel<<<g1((long)NE * UDIM), blk, 0, stream>>>(eattr, e0w, e0b, wbuf);

    for (int i = 0; i < NDEPTH; ++i) {
        const size_t oU = (size_t)i * UDIM;
        // weight conversion f32 -> f16
        gat_cvt_f16_kernel<<<g1(UDIM * UDIM), blk, 0, stream>>>(Wl + (size_t)i * UDIM * UDIM, Wl_h, UDIM * UDIM);
        gat_cvt_f16_kernel<<<g1(UDIM * UDIM), blk, 0, stream>>>(Wr + (size_t)i * UDIM * UDIM, Wr_h, UDIM * UDIM);
        gat_cvt_f16_kernel<<<g1(UDIM * UDIM), blk, 0, stream>>>(We + (size_t)i * UDIM * UDIM, We_h, UDIM * UDIM);
        gat_cvt_f16_kernel<<<g1(3 * UDIM * UDIM), blk, 0, stream>>>(W1 + (size_t)i * 3 * UDIM * UDIM, W1_h, 3 * UDIM * UDIM);
        gat_cvt_f16_kernel<<<g1(UDIM * UDIM), blk, 0, stream>>>(W2 + (size_t)i * UDIM * UDIM, W2_h, UDIM * UDIM);

        // clears / inits
        gat_zero_kernel<<<g1(NN), blk, 0, stream>>>(deg, NN);
        gat_zero_kernel<<<g1((long)NN * UDIM), blk, 0, stream>>>(lpat, (long)NN * UDIM);
        gat_zero_kernel<<<g1(NN * 4), blk, 0, stream>>>((float*)mkey, NN * 4);
        gat_zero_kernel<<<g1(NN * 4), blk, 0, stream>>>(ssum, NN * 4);
        gat_init_bias_kernel<<<g1((long)NN * UDIM), blk, 0, stream>>>(xnew, gbias + oU, NN);
        gat_zero_kernel<<<g1(256), blk, 0, stream>>>(bns, 256);

        // self-loop attr = segment_mean(w, col)
        gat_deg_kernel<<<g1(NE), blk, 0, stream>>>(coli, deg);
        gat_loop_acc_kernel<<<g1((long)NE * UDIM), blk, 0, stream>>>(wbuf, coli, lpat);
        gat_loop_div_kernel<<<g1((long)NN * UDIM), blk, 0, stream>>>(lpat, deg);

        // WMMA GEMMs
        gat_gemm128_kernel<0, true ><<<gNodeG, blk, 0, stream>>>(h, Wl_h, bl + oU, xl, NN);
        gat_gemm128_kernel<0, true ><<<gNodeG, blk, 0, stream>>>(h, Wr_h, br + oU, xr, NN);
        gat_gemm128_kernel<0, false><<<gEdgeG, blk, 0, stream>>>(wbuf, We_h, nullptr, bufA, NE);
        gat_gemm128_kernel<0, false><<<gNodeG, blk, 0, stream>>>(lpat, We_h, nullptr, bufA + (size_t)NE * UDIM, NN);

        // segment softmax attention
        gat_attn_score_kernel<<<g1((long)NR * 4), blk, 0, stream>>>(xl, xr, bufA, rowi, coli, att + oU, esc, mkey);
        gat_attn_exp_kernel<<<g1((long)NR * 4), blk, 0, stream>>>(esc, mkey, ssum, rowi, coli);
        gat_attn_agg_kernel<<<g1((long)NR * UDIM), blk, 0, stream>>>(esc, ssum, xl, rowi, coli, xnew);

        // node BN + silu + residual
        gat_bn_stats_kernel<<<dim3(512), blk, 0, stream>>>(xnew, bns, NN);
        gat_bn_apply_kernel<<<g1((long)NN * UDIM), blk, 0, stream>>>(xnew, bns, vbng + oU, vbnb + oU, h, NN);

        // edge MLP (WMMA)
        gat_edge_mlp1_kernel<<<gEdgeG, blk, 0, stream>>>(h, wbuf, rowi, coli, W1_h, b1 + oU, bufA);
        gat_gemm128_kernel<0, true><<<gEdgeG, blk, 0, stream>>>(bufA, W2_h, b2 + oU, bufB, NE);

        // edge BN + silu + residual
        gat_zero_kernel<<<g1(256), blk, 0, stream>>>(bns, 256);
        gat_bn_stats_kernel<<<dim3(512), blk, 0, stream>>>(bufB, bns, NE);
        gat_bn_apply_kernel<<<g1((long)NE * UDIM), blk, 0, stream>>>(bufB, bns, ebng + oU, ebnb + oU, wbuf, NE);
    }

    gat_copy_kernel<<<g1((long)NE * UDIM), blk, 0, stream>>>(wbuf, (float*)d_out, (long)NE * UDIM);
}
